// TeamWinPredictor_25658134627030
// MI455X (gfx1250) — matrix operand order repaired and checked by reading, not compiled, not run on hardware
//
#include <hip/hip_runtime.h>
#include <math.h>

#define NNODES 50000
#define NEDGES 640000
#define DFEAT  128
#define HFEAT  128
#define NGRAPH 64
#define LN_EPS 1e-5f

typedef __attribute__((ext_vector_type(16))) _Float16 v16h;
typedef __attribute__((ext_vector_type(8)))  _Float16 v8h;
typedef __attribute__((ext_vector_type(8)))  float    v8f;

// ---------------- utility kernels ----------------

__global__ void k_fill(float* __restrict__ p, long n, float v) {
  long i = (long)blockIdx.x * blockDim.x + threadIdx.x;
  long stride = (long)gridDim.x * blockDim.x;
  for (; i < n; i += stride) p[i] = v;
}

__global__ void k_deg_accum(const long long* __restrict__ col, float* __restrict__ deg, int ne) {
  int e = blockIdx.x * blockDim.x + threadIdx.x;
  if (e < ne) atomicAdd(&deg[(int)col[e]], 1.0f);
}

__global__ void k_deg_inv(float* __restrict__ deg, int n) {
  int i = blockIdx.x * blockDim.x + threadIdx.x;
  if (i < n) deg[i] = rsqrtf(deg[i]);   // deg >= 1 (self-loop included)
}

__global__ void k_f32_to_f16(const float* __restrict__ src, _Float16* __restrict__ dst, long n) {
  long i = (long)blockIdx.x * blockDim.x + threadIdx.x;
  long stride = (long)gridDim.x * blockDim.x;
  for (; i < n; i += stride) dst[i] = (_Float16)src[i];
}

// Pack W[128][128] (row-major, k-major rows) into per-lane B-fragment layout:
// wp[colTile(8)][kstep(4)][lane(32)][h(16)], lane holds N=lane&15,
// K = kstep*32 + (h&7) + (h>=8 ? 16 : 0) + (lane>=16 ? 8 : 0).
__global__ void k_pack_w(const float* __restrict__ W, _Float16* __restrict__ wp) {
  int idx = blockIdx.x * blockDim.x + threadIdx.x;   // 16384 total
  int h     = idx & 15;
  int lane  = (idx >> 4) & 31;
  int kstep = (idx >> 9) & 3;
  int tile  = idx >> 11;
  int ncol  = tile * 16 + (lane & 15);
  int k     = kstep * 32 + (h & 7) + ((h >> 3) << 4) + ((lane >= 16) ? 8 : 0);
  wp[idx] = (_Float16)W[k * HFEAT + ncol];
}

// ---------------- WMMA GEMM: out[rows x 128] = A[rows x 128] @ W ----------------
// One wave per 16x16 output tile, K=128 via 4x v_wmma_f32_16x16x32_f16.

__global__ __launch_bounds__(256) void k_gemm_wmma(const _Float16* __restrict__ A,
                                                   const _Float16* __restrict__ wp,
                                                   float* __restrict__ out, int rows) {
  const int wave = threadIdx.x >> 5;        // 8 waves = 8 column tiles
  const int lane = threadIdx.x & 31;
  const int rowBase = blockIdx.x * 16;
  if (rowBase >= rows) return;

  // A fragment addressing (16-bit A 16x32 layout):
  // lane holds M = lane&15; halves 0..7 -> K = k0 + koff .. +7,
  // halves 8..15 -> K = k0+16+koff .. +7, koff = (lane>=16)*8.
  const int m    = lane & 15;
  const int koff = (lane >> 4) << 3;
  const _Float16* arow = A + (long)(rowBase + m) * DFEAT;

  v8f acc = {};
#pragma unroll
  for (int ks = 0; ks < 4; ++ks) {
    const _Float16* ap = arow + ks * 32;
    v8h lo = *(const v8h*)(ap + koff);
    v8h hi = *(const v8h*)(ap + 16 + koff);
    v16h a;
#pragma unroll
    for (int i = 0; i < 8; ++i) { a[i] = lo[i]; a[i + 8] = hi[i]; }
    v16h b = *(const v16h*)(wp + (((long)wave * 4 + ks) * 32 + lane) * 16);
    acc = __builtin_amdgcn_wmma_f32_16x16x32_f16(false, a, false, b,
                                                 (short)0, acc, false, false);
  }

  // C layout: VGPR r holds (M = r + (lane>=16 ? 8 : 0), N = lane&15)
  const int colBase = wave * 16;
  float* op = out + (long)(rowBase + ((lane >> 4) << 3)) * HFEAT + colBase + (lane & 15);
#pragma unroll
  for (int r = 0; r < 8; ++r) op[(long)r * HFEAT] = acc[r];
}

// ---------------- edge aggregation: out[c] += h[r] * dinv[r]*dinv[c] ----------------
// One wave per edge (incl. implicit self-loops), each lane does float4.

__global__ __launch_bounds__(256) void k_aggregate(const long long* __restrict__ ei,
                                                   const float* __restrict__ dinv,
                                                   const float* __restrict__ h,
                                                   float* __restrict__ out,
                                                   int ne, int n) {
  const int t    = (blockIdx.x * blockDim.x + threadIdx.x) >> 5;
  const int lane = threadIdx.x & 31;
  if (t >= ne + n) return;
  int r, c;
  if (t < ne) { r = (int)ei[t]; c = (int)ei[ne + t]; }
  else        { r = t - ne;     c = r; }
  const float nrm = dinv[r] * dinv[c];
  const float4 v = *(const float4*)(h + (long)r * HFEAT + lane * 4);
  float* o = out + (long)c * HFEAT + lane * 4;
  atomicAdd(o + 0, v.x * nrm);
  atomicAdd(o + 1, v.y * nrm);
  atomicAdd(o + 2, v.z * nrm);
  atomicAdd(o + 3, v.w * nrm);
}

// ---------------- bias add + global sum/sumsq ----------------

__global__ void k_bias_stats(float* __restrict__ x, const float* __restrict__ bias,
                             float* __restrict__ stats, long n) {
  long i = (long)blockIdx.x * blockDim.x + threadIdx.x;
  long stride = (long)gridDim.x * blockDim.x;
  float s = 0.f, s2 = 0.f;
  for (; i < n; i += stride) {
    float v = x[i] + bias[i & (HFEAT - 1)];
    x[i] = v;
    s += v; s2 += v * v;
  }
  for (int off = 16; off > 0; off >>= 1) {
    s  += __shfl_xor(s,  off, 32);
    s2 += __shfl_xor(s2, off, 32);
  }
  if ((threadIdx.x & 31) == 0) { atomicAdd(&stats[0], s); atomicAdd(&stats[1], s2); }
}

__global__ void k_stats_final(float* __restrict__ stats, float count) {
  float mu  = stats[0] / count;
  float var = stats[1] / count - mu * mu;
  float sd  = sqrtf(fmaxf(var, 0.f));
  stats[2] = mu;
  stats[3] = 1.0f / (sd + LN_EPS);
}

__global__ void k_ln_relu(const float* __restrict__ x, const float* __restrict__ stats,
                          const float* __restrict__ w, const float* __restrict__ b,
                          float* __restrict__ y32, _Float16* __restrict__ y16, long n) {
  const float mu  = stats[2];
  const float inv = stats[3];
  long i = (long)blockIdx.x * blockDim.x + threadIdx.x;
  long stride = (long)gridDim.x * blockDim.x;
  for (; i < n; i += stride) {
    int j = (int)(i & (HFEAT - 1));
    float v = (x[i] - mu) * inv * w[j] + b[j];
    v = fmaxf(v, 0.f);
    if (y32) y32[i] = v;
    if (y16) y16[i] = (_Float16)v;
  }
}

// ---------------- pooling + head ----------------

__global__ __launch_bounds__(256) void k_pool(const float* __restrict__ h,
                                              const long long* __restrict__ batch,
                                              float* __restrict__ pooled,
                                              float* __restrict__ cnt, int n) {
  const int t    = (blockIdx.x * blockDim.x + threadIdx.x) >> 5;
  const int lane = threadIdx.x & 31;
  if (t >= n) return;
  const int g = (int)batch[t];
  const float4 v = *(const float4*)(h + (long)t * HFEAT + lane * 4);
  float* o = pooled + (long)g * HFEAT + lane * 4;
  atomicAdd(o + 0, v.x); atomicAdd(o + 1, v.y);
  atomicAdd(o + 2, v.z); atomicAdd(o + 3, v.w);
  if (lane == 0) atomicAdd(&cnt[g], 1.0f);
}

__global__ __launch_bounds__(128) void k_head(const float* __restrict__ pooled,
                                              const float* __restrict__ cnt,
                                              const float* __restrict__ fcw,
                                              const float* __restrict__ fcb,
                                              float* __restrict__ out) {
  __shared__ float sdata[128];
  const int g = blockIdx.x;
  const int t = threadIdx.x;
  const float c = fmaxf(cnt[g], 1.0f);
  sdata[t] = pooled[g * HFEAT + t] / c * fcw[t];
  __syncthreads();
  for (int off = 64; off > 0; off >>= 1) {
    if (t < off) sdata[t] += sdata[t + off];
    __syncthreads();
  }
  if (t == 0) {
    float z = sdata[0] + fcb[0];
    out[g] = 1.0f / (1.0f + expf(-z));
  }
}

// ---------------- host orchestration ----------------

extern "C" void kernel_launch(void* const* d_in, const int* in_sizes, int n_in,
                              void* d_out, int out_size, void* d_ws, size_t ws_size,
                              hipStream_t stream) {
  (void)in_sizes; (void)n_in; (void)out_size; (void)ws_size;
  const float*     x     = (const float*)d_in[0];
  const long long* ei    = (const long long*)d_in[1];   // [2,E] int64
  const long long* batch = (const long long*)d_in[2];   // [N] int64
  const float* W1   = (const float*)d_in[3];
  const float* b1   = (const float*)d_in[4];
  const float* ln1w = (const float*)d_in[5];
  const float* ln1b = (const float*)d_in[6];
  const float* W2   = (const float*)d_in[7];
  const float* b2   = (const float*)d_in[8];
  const float* ln2w = (const float*)d_in[9];
  const float* ln2b = (const float*)d_in[10];
  const float* fcw  = (const float*)d_in[11];
  const float* fcb  = (const float*)d_in[12];
  float* out = (float*)d_out;

  char* ws = (char*)d_ws;
  size_t off = 0;
  auto alloc = [&](size_t bytes) -> void* {
    void* p = ws + off;
    off = (off + bytes + 255) & ~(size_t)255;
    return p;
  };

  float*    dinv   = (float*)alloc((size_t)NNODES * 4);
  float*    stats  = (float*)alloc(4 * 4);
  float*    pooled = (float*)alloc((size_t)(NGRAPH * HFEAT + NGRAPH) * 4);
  float*    cnt    = pooled + NGRAPH * HFEAT;
  _Float16* wp     = (_Float16*)alloc((size_t)HFEAT * HFEAT * 2);
  _Float16* xh     = (_Float16*)alloc((size_t)NNODES * HFEAT * 2);
  float*    hlin   = (float*)alloc((size_t)NNODES * HFEAT * 4);
  float*    hagg   = (float*)alloc((size_t)NNODES * HFEAT * 4);

  const long NH = (long)NNODES * HFEAT;
  const int edgeWaves = NEDGES + NNODES;                     // edges + self-loops
  const int aggBlocks = (edgeWaves * 32 + 255) / 256;
  const int poolBlocks = (NNODES * 32 + 255) / 256;
  const int gemmBlocks = NNODES / 16;                        // 3125

  // degrees -> dinv (self-loop gives deg>=1)
  k_fill<<<256, 256, 0, stream>>>(dinv, NNODES, 1.0f);
  k_deg_accum<<<(NEDGES + 255) / 256, 256, 0, stream>>>(ei + NEDGES, dinv, NEDGES);
  k_deg_inv<<<(NNODES + 255) / 256, 256, 0, stream>>>(dinv, NNODES);

  // ---- layer 1 ----
  k_f32_to_f16<<<2048, 256, 0, stream>>>(x, xh, NH);
  k_pack_w<<<64, 256, 0, stream>>>(W1, wp);
  k_gemm_wmma<<<gemmBlocks, 256, 0, stream>>>(xh, wp, hlin, NNODES);
  k_fill<<<2048, 256, 0, stream>>>(hagg, NH, 0.0f);
  k_aggregate<<<aggBlocks, 256, 0, stream>>>(ei, dinv, hlin, hagg, NEDGES, NNODES);
  k_fill<<<1, 32, 0, stream>>>(stats, 4, 0.0f);
  k_bias_stats<<<1024, 256, 0, stream>>>(hagg, b1, stats, NH);
  k_stats_final<<<1, 1, 0, stream>>>(stats, (float)NH);
  k_ln_relu<<<2048, 256, 0, stream>>>(hagg, stats, ln1w, ln1b, nullptr, xh, NH);

  // ---- layer 2 ----
  k_pack_w<<<64, 256, 0, stream>>>(W2, wp);
  k_gemm_wmma<<<gemmBlocks, 256, 0, stream>>>(xh, wp, hlin, NNODES);
  k_fill<<<2048, 256, 0, stream>>>(hagg, NH, 0.0f);
  k_aggregate<<<aggBlocks, 256, 0, stream>>>(ei, dinv, hlin, hagg, NEDGES, NNODES);
  k_fill<<<1, 32, 0, stream>>>(stats, 4, 0.0f);
  k_bias_stats<<<1024, 256, 0, stream>>>(hagg, b2, stats, NH);
  k_stats_final<<<1, 1, 0, stream>>>(stats, (float)NH);
  k_ln_relu<<<2048, 256, 0, stream>>>(hagg, stats, ln2w, ln2b, hlin, nullptr, NH);

  // ---- pool + head ----
  k_fill<<<64, 256, 0, stream>>>(pooled, NGRAPH * HFEAT + NGRAPH, 0.0f);
  k_pool<<<poolBlocks, 256, 0, stream>>>(hlin, batch, pooled, cnt, NNODES);
  k_head<<<NGRAPH, 128, 0, stream>>>(pooled, cnt, fcw, fcb, out);
}
